// LSTM_33397665693860
// MI455X (gfx1250) — matrix operand order repaired and checked by reading, not compiled, not run on hardware
//
#include <hip/hip_runtime.h>
#include <hip/hip_bf16.h>

// Problem dims (fixed by reference)
#define NB 64          // batch
#define TT 512         // time steps
#define DD 512         // input dim
#define HH 512         // hidden dim
#define G4 2048        // 4*H
#define NT (NB*TT)     // 32768 rows of x

#define NWG 32         // workgroups in the persistent recurrent kernel

typedef __attribute__((ext_vector_type(16))) __bf16 v16bf;
typedef __attribute__((ext_vector_type(8)))  float  v8f;
typedef __attribute__((ext_vector_type(4)))  unsigned u32x4;
typedef __attribute__((ext_vector_type(8)))  int      i32x8;
typedef __attribute__((ext_vector_type(4)))  int      i32x4;

#if defined(__has_builtin)
#if __has_builtin(__builtin_amdgcn_tensor_load_to_lds)
#define HAVE_TDM 1
#endif
#endif

struct U16x16 { u32x4 lo, hi; };
static_assert(sizeof(U16x16) == 32, "frag size");

__device__ __forceinline__ v16bf make_frag(u32x4 lo, u32x4 hi) {
    U16x16 t{lo, hi};
    return __builtin_bit_cast(v16bf, t);
}

__device__ __forceinline__ float sigmoid_fast(float x) {
    return 1.0f / (1.0f + __expf(-x));
}
__device__ __forceinline__ float tanh_fast(float x) {
    // tanh(x) = 1 - 2/(exp(2x)+1)
    return 1.0f - 2.0f / (1.0f + __expf(2.0f * x));
}

// ---------------------------------------------------------------------------
// Prep kernels: f32 -> bf16 conversion + weight transposes + state init
// ---------------------------------------------------------------------------
__global__ void prep_x_bf16(const float* __restrict__ x,
                            __hip_bfloat16* __restrict__ xb, int n) {
    int stride = gridDim.x * blockDim.x;
    for (int i = blockIdx.x * blockDim.x + threadIdx.x; i < n; i += stride)
        xb[i] = __float2bfloat16(x[i]);
}

// WxT[col*DD + k] = Wx[k*G4 + col]; same for WhT (col-major bf16 weights)
__global__ void prep_w_transpose(const float* __restrict__ Wx,
                                 const float* __restrict__ Wh,
                                 __hip_bfloat16* __restrict__ WxT,
                                 __hip_bfloat16* __restrict__ WhT) {
    const int total = G4 * DD;           // per matrix
    int stride = gridDim.x * blockDim.x;
    for (int i = blockIdx.x * blockDim.x + threadIdx.x; i < 2 * total; i += stride) {
        if (i < total) {
            int col = i >> 9, k = i & (DD - 1);
            WxT[i] = __float2bfloat16(Wx[(size_t)k * G4 + col]);
        } else {
            int j = i - total;
            int col = j >> 9, k = j & (HH - 1);
            WhT[j] = __float2bfloat16(Wh[(size_t)k * G4 + col]);
        }
    }
}

// hbuf[0] <- bf16(h0); zero the global barrier state
__global__ void prep_state(const float* __restrict__ h0,
                           __hip_bfloat16* __restrict__ hbuf,
                           unsigned* __restrict__ sync) {
    int i = blockIdx.x * blockDim.x + threadIdx.x;   // exactly NB*HH threads
    if (i < NB * HH) hbuf[i] = __float2bfloat16(h0[i]);
    if (i == 0) { sync[0] = 0u; sync[1] = 0u; }
}

// ---------------------------------------------------------------------------
// Big GEMM: xW_bf16 = bf16( x_bf @ Wx + b )   (32768 x 2048, K = 512)
// One wave computes a 2x4 block of 16x16 tiles (32 rows x 64 cols): 4x A-reuse
// and 2x B-reuse in registers -> ~3x less L2 traffic than 1 tile/wave.
// ---------------------------------------------------------------------------
__global__ void gemm_xw_bf16(const __hip_bfloat16* __restrict__ xb,
                             const __hip_bfloat16* __restrict__ WxT,
                             const float* __restrict__ bias,
                             __hip_bfloat16* __restrict__ xWb) {
    const int lane = threadIdx.x & 31;
    const int wid  = (blockIdx.x << 3) + (threadIdx.x >> 5);  // 32768 wave-blocks
    const int wn   = wid & 31;                  // 32 col-groups of 64
    const int wm   = wid >> 5;                  // 1024 row-groups of 32
    const int l15  = lane & 15;
    const int half = lane >> 4;

    const int r0 = wm * 32 + l15;               // rows r0, r0+16
    const int c0 = wn * 64 + l15;               // cols c0 + 16*j, j=0..3

    const u32x4* ap[2];
    ap[0] = (const u32x4*)(xb + (size_t)r0 * DD);
    ap[1] = (const u32x4*)(xb + (size_t)(r0 + 16) * DD);
    const u32x4* bp[4];
#pragma unroll
    for (int j = 0; j < 4; ++j)
        bp[j] = (const u32x4*)(WxT + (size_t)(c0 + 16 * j) * DD);

    v8f acc[2][4];
#pragma unroll
    for (int i = 0; i < 2; ++i)
#pragma unroll
        for (int j = 0; j < 4; ++j)
            acc[i][j] = (v8f){0.f,0.f,0.f,0.f,0.f,0.f,0.f,0.f};

#pragma unroll 2
    for (int kk = 0; kk < 16; ++kk) {
        int ia = kk * 4 + half * 2;             // A: lane half h -> 16 contig K @16h
        int ib = kk * 4 + half * 2;             // B: lane half h -> 16 contig K @16h
        v16bf A[2], B[4];
#pragma unroll
        for (int i = 0; i < 2; ++i) A[i] = make_frag(ap[i][ia], ap[i][ia + 1]);
#pragma unroll
        for (int j = 0; j < 4; ++j) B[j] = make_frag(bp[j][ib], bp[j][ib + 1]);
#pragma unroll
        for (int i = 0; i < 2; ++i)
#pragma unroll
            for (int j = 0; j < 4; ++j)
                acc[i][j] = __builtin_amdgcn_wmma_f32_16x16x32_bf16(
                                false, A[i], false, B[j], (short)0, acc[i][j],
                                false, false);
    }

#pragma unroll
    for (int j = 0; j < 4; ++j) {
        const int cn = c0 + 16 * j;
        const float bv = bias[cn];
#pragma unroll
        for (int i = 0; i < 2; ++i) {
            const int rbase = wm * 32 + 16 * i + half * 8;  // C/D: M = e + 8*half
#pragma unroll
            for (int e = 0; e < 8; ++e) {
                int row = rbase + e;
                xWb[(size_t)row * G4 + cn] = __float2bfloat16(acc[i][j][e] + bv);
            }
        }
    }
}

// ---------------------------------------------------------------------------
// Persistent recurrent kernel: 32 WGs x 128 threads.
// Block hb owns h-columns [hb*16, hb*16+16); wave w owns batch rows
// [w*16, w*16+16). Each wave computes all 4 gate tiles -> c/h stay in
// registers; Wh slice (4 gates x 16 cols x 512 K, bf16) staged into 64KB LDS
// once via the Tensor Data Mover (TDM). Cross-WG sync per step via a global
// generation barrier.
// ---------------------------------------------------------------------------
__global__ void lstm_recurrent(const __hip_bfloat16* __restrict__ xWb,
                               const __hip_bfloat16* __restrict__ WhT,
                               __hip_bfloat16* __restrict__ hbuf,   // [2][NB*HH]
                               float* __restrict__ out,             // (N,T,H)
                               unsigned* __restrict__ sync) {       // {count, gen}
    __shared__ __align__(16) __hip_bfloat16 Bsh[4 * 16 * HH];       // 64 KB

    const int hb   = blockIdx.x;            // 0..31
    const int lane = threadIdx.x & 31;
    const int w    = threadIdx.x >> 5;      // 0..3  (batch row tile)
    const int l15  = lane & 15;
    const int half = lane >> 4;

    // ---- stage Wh slice into LDS (once) --------------------------------
    // 3D TDM tile: x = one K-row (128 x 8B = 512 bf16), y = 16 cols of a
    // gate block, z = 4 gate blocks strided 512 rows apart. Lands
    // contiguously in LDS as Bsh[g][cl][k].
#if HAVE_TDM
    if (threadIdx.x < 32) {                  // one wave issues the DMA
        unsigned lds_addr = (unsigned)(uintptr_t)(&Bsh[0]);
        unsigned long long ga =
            (unsigned long long)(uintptr_t)(WhT + (size_t)(hb * 16) * HH);
        u32x4 g0 = { 1u,                                    // count=1 (valid D#)
                     lds_addr,                              // lds_addr
                     (unsigned)(ga & 0xFFFFFFFFu),          // global_addr[31:0]
                     (unsigned)((ga >> 32) & 0x01FFFFFFu)   // global_addr[56:32]
                         | (2u << 30) };                    // type=2 ("image")
        i32x8 g1 = { (int)0x00030000,       // wg_mask=0, data_size=3 (8B)
                     (int)(128u << 16),     // tensor_dim0[15:0]=128 @ [31:16]
                     (int)(2048u << 16),    // tensor_dim1[15:0]=2048 @ [31:16]
                     (int)(128u << 16),     // tile_dim0=128 @ [31:16]
                     (int)((4u << 16) | 16u), // tile_dim1=16, tile_dim2=4
                     (int)128,              // tensor_dim0_stride = 128 (1 row)
                     (int)0,                // stride0 hi / stride1 lo
                     (int)1 };              // tensor_dim1_stride = 65536 (512 rows)
        i32x4 g2 = { 4, 1, 0, 0 };          // tensor_dim2=4, dim3=1, stride2=0, tile_dim3=0
        i32x4 g3 = { 0, 0, 0, 0 };
#if defined(__clang_major__) && __clang_major__ >= 23
        i32x8 g4 = { 0, 0, 0, 0, 0, 0, 0, 0 };
        __builtin_amdgcn_tensor_load_to_lds(g0, g1, g2, g3, g4, 0);
#else
        __builtin_amdgcn_tensor_load_to_lds(g0, g1, g2, g3, 0);
#endif
        __builtin_amdgcn_s_wait_tensorcnt(0);
    }
#else
    {
        const int nu4 = (4 * 16 * HH * 2) / 16;      // 4096 u32x4
        u32x4* dst = (u32x4*)Bsh;
        for (int i = threadIdx.x; i < nu4; i += blockDim.x) {
            int lc = i >> 6;                          // local col = g*16 + cl
            int within = i & 63;
            int g = lc >> 4, cl = lc & 15;
            const u32x4* src =
                (const u32x4*)(WhT + (size_t)(g * HH + hb * 16 + cl) * HH);
            dst[i] = src[within];
        }
    }
#endif
    __syncthreads();

    const int r  = w * 16 + l15;            // batch row for A-fragment lane
    const int hc = hb * 16 + l15;           // global h column for this lane
    const int rbase = w * 16 + half * 8;    // C/D layout: M = e + 8*half
    const u32x4* bbase = (const u32x4*)Bsh;

    v8f cacc = {0.f, 0.f, 0.f, 0.f, 0.f, 0.f, 0.f, 0.f};  // c state in regs

    for (int t = 0; t < TT; ++t) {
        const __hip_bfloat16* hprev = hbuf + (size_t)(t & 1) * (NB * HH);
        __hip_bfloat16* hnext       = hbuf + (size_t)((t + 1) & 1) * (NB * HH);

        // Prefetch next step's xW gate rows (hides the only HBM latency in
        // the serial chain behind this step's GEMM).
        if (t + 1 < TT) {
            const __hip_bfloat16* pf =
                xWb + ((size_t)rbase * TT + (t + 1)) * G4 + hc;
#pragma unroll
            for (int g = 0; g < 4; ++g)
                __builtin_prefetch(pf + g * HH, 0, 1);
        }

        v8f acc0 = {0.f,0.f,0.f,0.f,0.f,0.f,0.f,0.f};
        v8f acc1 = acc0, acc2 = acc0, acc3 = acc0;

        const u32x4* arow = (const u32x4*)(hprev + (size_t)r * HH);
#pragma unroll 4
        for (int kk = 0; kk < 16; ++kk) {
            int ia = kk * 4 + half * 2;
            v16bf A = make_frag(arow[ia], arow[ia + 1]);
            int ib = kk * 4 + half * 2;
            // gate g column block lives at LDS row (g*16 + l15)
            v16bf B0 = make_frag(bbase[(0*16 + l15)*64 + ib], bbase[(0*16 + l15)*64 + ib + 1]);
            v16bf B1 = make_frag(bbase[(1*16 + l15)*64 + ib], bbase[(1*16 + l15)*64 + ib + 1]);
            v16bf B2 = make_frag(bbase[(2*16 + l15)*64 + ib], bbase[(2*16 + l15)*64 + ib + 1]);
            v16bf B3 = make_frag(bbase[(3*16 + l15)*64 + ib], bbase[(3*16 + l15)*64 + ib + 1]);
            acc0 = __builtin_amdgcn_wmma_f32_16x16x32_bf16(false, A, false, B0, (short)0, acc0, false, false);
            acc1 = __builtin_amdgcn_wmma_f32_16x16x32_bf16(false, A, false, B1, (short)0, acc1, false, false);
            acc2 = __builtin_amdgcn_wmma_f32_16x16x32_bf16(false, A, false, B2, (short)0, acc2, false, false);
            acc3 = __builtin_amdgcn_wmma_f32_16x16x32_bf16(false, A, false, B3, (short)0, acc3, false, false);
        }

        // ---- gates + state update (all in-register per lane) -----------
#pragma unroll
        for (int e = 0; e < 8; ++e) {
            int n = rbase + e;
            size_t xwrow = ((size_t)n * TT + t) * G4;
            float ai = acc0[e] + __bfloat162float(xWb[xwrow + 0 * HH + hc]);
            float af = acc1[e] + __bfloat162float(xWb[xwrow + 1 * HH + hc]);
            float ao = acc2[e] + __bfloat162float(xWb[xwrow + 2 * HH + hc]);
            float ag = acc3[e] + __bfloat162float(xWb[xwrow + 3 * HH + hc]);
            float ig = sigmoid_fast(ai);
            float fg = sigmoid_fast(af);
            float og = sigmoid_fast(ao);
            float gg = tanh_fast(ag);
            float c  = fg * cacc[e] + ig * gg;
            cacc[e]  = c;
            float h  = og * tanh_fast(c);
            out[((size_t)n * TT + t) * HH + hc] = h;
            hnext[(size_t)n * HH + hc] = __float2bfloat16(h);
        }

        // ---- device-wide step barrier ----------------------------------
        if (t != TT - 1) {
            __threadfence();        // make hnext visible at agent scope
            __syncthreads();
            if (threadIdx.x == 0) {
                unsigned old = __hip_atomic_fetch_add(&sync[0], 1u,
                                   __ATOMIC_ACQ_REL, __HIP_MEMORY_SCOPE_AGENT);
                if (old == NWG - 1) {
                    __hip_atomic_store(&sync[0], 0u,
                                       __ATOMIC_RELAXED, __HIP_MEMORY_SCOPE_AGENT);
                    __hip_atomic_store(&sync[1], (unsigned)(t + 1),
                                       __ATOMIC_RELEASE, __HIP_MEMORY_SCOPE_AGENT);
                } else {
                    while (__hip_atomic_load(&sync[1], __ATOMIC_ACQUIRE,
                                             __HIP_MEMORY_SCOPE_AGENT) < (unsigned)(t + 1))
                        __builtin_amdgcn_s_sleep(1);
                }
            }
            __syncthreads();
        }
    }
}

// ---------------------------------------------------------------------------
extern "C" void kernel_launch(void* const* d_in, const int* in_sizes, int n_in,
                              void* d_out, int out_size, void* d_ws, size_t ws_size,
                              hipStream_t stream) {
    (void)in_sizes; (void)n_in; (void)out_size; (void)ws_size;
    const float* x    = (const float*)d_in[0];
    const float* h0   = (const float*)d_in[1];
    const float* Wx   = (const float*)d_in[2];
    const float* Wh   = (const float*)d_in[3];
    const float* bias = (const float*)d_in[4];
    float* out = (float*)d_out;

    // Workspace layout (bytes):
    char* ws = (char*)d_ws;
    __hip_bfloat16* xWb  = (__hip_bfloat16*)(ws);                    // 32768*2048*2 = 134217728
    __hip_bfloat16* xb   = (__hip_bfloat16*)(ws + 134217728);        // 32768*512*2  = 33554432
    __hip_bfloat16* WxT  = (__hip_bfloat16*)(ws + 167772160);        // 2048*512*2   = 2097152
    __hip_bfloat16* WhT  = (__hip_bfloat16*)(ws + 169869312);        // 2048*512*2   = 2097152
    __hip_bfloat16* hbuf = (__hip_bfloat16*)(ws + 171966464);        // 2*64*512*2   = 131072
    unsigned*       sync = (unsigned*)      (ws + 172097536);        // 2*u32

    prep_x_bf16     <<<8192, 256, 0, stream>>>(x, xb, NT * DD);
    prep_w_transpose<<<2048, 256, 0, stream>>>(Wx, Wh, WxT, WhT);
    prep_state      <<<128,  256, 0, stream>>>(h0, hbuf, sync);
    gemm_xw_bf16    <<<4096, 256, 0, stream>>>(xb, WxT, bias, xWb);
    lstm_recurrent  <<<NWG,  128, 0, stream>>>(xWb, WhT, hbuf, out, sync);
}
